// LSTMClassifier_65601330479134
// MI455X (gfx1250) — matrix operand order repaired and checked by reading, not compiled, not run on hardware
//
#include <hip/hip_runtime.h>
#include <hip/hip_bf16.h>

#define B_  256
#define L_  2048
#define V_  24
#define H_  256
#define C_  10
#define G4H 1024  // 4*H

typedef __attribute__((ext_vector_type(16))) __bf16 v16bf;
typedef __attribute__((ext_vector_type(8)))  float  v8f;

union AFrag { v16bf v; uint4 q[2]; };

// workspace layout (bytes)
#define WS_ORDER 0
#define WS_E     (B_ * 4)                    // 24*1024 f32 = 96KB
#define WS_WB    (WS_E + V_ * G4H * 4)       // 256*1024 bf16 = 512KB, fragment-swizzled

#if defined(__HIP_DEVICE_COMPILE__) && __has_builtin(__builtin_amdgcn_tanhf)
__device__ __forceinline__ float dev_tanh(float x) { return __builtin_amdgcn_tanhf(x); }
#elif defined(__HIP_DEVICE_COMPILE__) && __has_builtin(__builtin_amdgcn_rcpf)
__device__ __forceinline__ float dev_tanh(float x) {
  float e = __expf(-2.0f * fabsf(x));
  float r = (1.0f - e) * __builtin_amdgcn_rcpf(1.0f + e);
  return copysignf(r, x);
}
#else
__device__ __forceinline__ float dev_tanh(float x) {
  float e = __expf(-2.0f * fabsf(x));
  float r = (1.0f - e) / (1.0f + e);
  return copysignf(r, x);
}
#endif

__device__ __forceinline__ float fast_sigmoid(float x) {
  // sigmoid(x) = 0.5*tanh(x/2) + 0.5  -> mul + v_tanh_f32 + fma, no div/exp
  return __builtin_fmaf(0.5f, dev_tanh(0.5f * x), 0.5f);
}

// Stable descending argsort by O(B^2) rank counting: order[rank] = i
__global__ void sort_kernel(const int* __restrict__ lens, int* __restrict__ order) {
  int i = threadIdx.x;
  int li = lens[i];
  int rank = 0;
  for (int j = 0; j < B_; ++j) {
    int lj = lens[j];
    rank += (lj > li) || (lj == li && j < i);
  }
  order[rank] = i;
}

// E[v, n] = W_ih[n, v] + b_ih[n] + b_hh[n]   (plain gate-block column order)
__global__ void prep_E(const float* __restrict__ Wih, const float* __restrict__ bih,
                       const float* __restrict__ bhh, float* __restrict__ E) {
  int v = blockIdx.x;      // 0..23
  int n = threadIdx.x;     // 0..1023
  E[v * G4H + n] = Wih[n * V_ + v] + bih[n] + bhh[n];
}

// Pre-swizzle W_hh^T into exact WMMA B-fragment lane order.
// Global tile ntg = w*8 + nt, nt = gate*2 + half:
//   column n = gate*256 + w*32 + half*16 + (L&15)   (row of W_hh = n)
// Fragment (ntg,kk): 32 lanes x 16 bf16; lane L local k = (L>>4)*16 + e; k = kk*32 + klocal.
__global__ void prep_WB(const float* __restrict__ Whh, __bf16* __restrict__ WB) {
  int flat = blockIdx.x * blockDim.x + threadIdx.x;  // 0..262143
  int e   = flat & 15;
  int Ln  = (flat >> 4) & 31;
  int kk  = (flat >> 9) & 7;
  int ntg = flat >> 12;
  int w   = ntg >> 3;
  int nt  = ntg & 7;
  int hi  = Ln >> 4;
  int k   = kk * 32 + hi * 16 + e;
  int n   = (nt >> 1) * H_ + w * 32 + (nt & 1) * 16 + (Ln & 15);
  WB[flat] = (__bf16)Whh[n * H_ + k];
}

__global__ __launch_bounds__(256) void lstm_kernel(
    const int* __restrict__ seqs, const int* __restrict__ lens,
    const int* __restrict__ order, const float* __restrict__ E,
    const __bf16* __restrict__ WB, const float* __restrict__ Wfc,
    const float* __restrict__ bfc, float* __restrict__ out) {
  // h as bf16, row stride 264 halves (528B = 33*16B: keeps b128 alignment, kills bank conflicts)
  __shared__ __bf16 h_bf[16 * 264];
  __shared__ float  h_s[16 * 256];
  __shared__ int    tok_s[16];
  __shared__ int    len_s[16];
  __shared__ int    ord_s[16];

  const int tid  = threadIdx.x;
  const int lane = tid & 31;
  const int w    = tid >> 5;          // wave 0..7: owns j in [32w, 32w+32) for all 4 gates
  const int hi   = (lane >> 4) & 1;
  const int r0   = blockIdx.x * 16;   // sorted-position base

  for (int i = tid; i < 16 * 264; i += 256) h_bf[i] = (__bf16)0.0f;
  if (tid < 16) {
    int o = order[r0 + tid];
    ord_s[tid] = o;
    len_s[tid] = lens[o];
    tok_s[tid] = seqs[o * L_];
  }
  __syncthreads();
  const int maxlen = len_s[0];  // globally sorted descending -> row 0 is block max

  // Hoist lengths + per-lane j into registers (masks are register compares in the loop)
  int lenm[8];
#pragma unroll
  for (int v = 0; v < 8; ++v) lenm[v] = len_s[v + 8 * hi];
  const int jlane = w * 32 + (lane & 15);   // + half*16

  // c and h state in registers: [half][v] for (m = v+8hi, j = jlane + 16*half)
  float creg[2][8], hreg[2][8];
#pragma unroll
  for (int hf = 0; hf < 2; ++hf)
#pragma unroll
    for (int v = 0; v < 8; ++v) { creg[hf][v] = 0.0f; hreg[hf][v] = 0.0f; }

  const char* hbytes = (const char*)h_bf;
  // Per-wave B-fragment base (bf16 elements): fragments (w*8+nt, kk) contiguous
  const __bf16* wbw = WB + ((w * 64) << 9) + lane * 16;

  for (int t = 0; t < maxlen; ++t) {
    int tk[8];
#pragma unroll
    for (int v = 0; v < 8; ++v) tk[v] = tok_s[v + 8 * hi];

    // acc[nt][v] = E[tok_m, n(nt)] : x-gates + folded biases;  nt = gate*2 + half
    v8f acc[8];
#pragma unroll
    for (int nt = 0; nt < 8; ++nt) {
      int n = (nt >> 1) * H_ + w * 32 + (nt & 1) * 16 + (lane & 15);
#pragma unroll
      for (int v = 0; v < 8; ++v) acc[nt][v] = E[tk[v] * G4H + n];
    }

    // g_tile += h_tile (16x256 bf16) @ W' slice; kk rolled to bound VGPR pressure
#pragma unroll 1
    for (int kk = 0; kk < 8; ++kk) {
      AFrag a;
      int ao = (lane & 15) * 528 + kk * 64 + hi * 32;
      a.q[0] = *(const uint4*)(hbytes + ao);
      a.q[1] = *(const uint4*)(hbytes + ao + 16);
#pragma unroll
      for (int nt = 0; nt < 8; ++nt) {
        AFrag b;
        const uint4* bp = (const uint4*)(wbw + ((nt * 8 + kk) << 9));
        b.q[0] = bp[0];
        b.q[1] = bp[1];
        acc[nt] = __builtin_amdgcn_wmma_f32_16x16x32_bf16(
            false, a.v, false, b.v, (short)0, acc[nt], false, false);
      }
    }

    __syncthreads();  // all waves done reading h_bf / tok_s for step t

    // Gates: i,f,g,o for (m,j) are in the SAME lane across acc[0|2|4|6 + half].
    // Branchless masked update, all 32 lanes productive, no shuffles, no div/exp.
#pragma unroll
    for (int hf = 0; hf < 2; ++hf) {
#pragma unroll
      for (int v = 0; v < 8; ++v) {
        float gi = acc[0 + hf][v];
        float gf = acc[2 + hf][v];
        float gg = acc[4 + hf][v];
        float go = acc[6 + hf][v];
        float c_old = creg[hf][v];
        float cn = fast_sigmoid(gf) * c_old + fast_sigmoid(gi) * dev_tanh(gg);
        float hn = fast_sigmoid(go) * dev_tanh(cn);
        bool  upd = t < lenm[v];
        cn = upd ? cn : c_old;
        hn = upd ? hn : hreg[hf][v];
        creg[hf][v] = cn;
        hreg[hf][v] = hn;
        int m = v + 8 * hi;
        int j = jlane + hf * 16;
        h_bf[m * 264 + j] = (__bf16)hn;
      }
    }

    if (tid < 16 && (t + 1) < maxlen) tok_s[tid] = seqs[ord_s[tid] * L_ + t + 1];
    __syncthreads();  // h_bf / tok_s for step t+1 visible
  }

  // Publish final f32 h once, then tiny FC (output stays in sorted order)
#pragma unroll
  for (int hf = 0; hf < 2; ++hf)
#pragma unroll
    for (int v = 0; v < 8; ++v)
      h_s[(v + 8 * hi) * 256 + jlane + hf * 16] = hreg[hf][v];
  __syncthreads();

  if (tid < 160) {
    int m = tid / C_, cls = tid % C_;
    float s = bfc[cls];
    for (int k = 0; k < H_; ++k) s += h_s[m * 256 + k] * Wfc[cls * H_ + k];
    out[(r0 + m) * C_ + cls] = s;
  }
}

extern "C" void kernel_launch(void* const* d_in, const int* in_sizes, int n_in,
                              void* d_out, int out_size, void* d_ws, size_t ws_size,
                              hipStream_t stream) {
  (void)in_sizes; (void)n_in; (void)out_size; (void)ws_size;
  const int*   seqs  = (const int*)d_in[0];
  const int*   lensp = (const int*)d_in[1];
  const float* Wih   = (const float*)d_in[2];
  const float* Whh   = (const float*)d_in[3];
  const float* bih   = (const float*)d_in[4];
  const float* bhh   = (const float*)d_in[5];
  const float* Wfc   = (const float*)d_in[6];
  const float* bfc   = (const float*)d_in[7];
  float*       out   = (float*)d_out;

  char*   ws    = (char*)d_ws;
  int*    order = (int*)(ws + WS_ORDER);
  float*  E     = (float*)(ws + WS_E);
  __bf16* WB    = (__bf16*)(ws + WS_WB);

  sort_kernel<<<1, B_, 0, stream>>>(lensp, order);
  prep_E<<<V_, G4H, 0, stream>>>(Wih, bih, bhh, E);
  prep_WB<<<512, 512, 0, stream>>>(Whh, WB);
  lstm_kernel<<<B_ / 16, 256, 0, stream>>>(seqs, lensp, order, E, WB, Wfc, bfc, out);
}
